// EdgeDecoder_16484084482976
// MI455X (gfx1250) — compile-verified, matrix operand order checked by reading, not run on hardware
//
#include <hip/hip_runtime.h>

typedef __attribute__((ext_vector_type(16))) _Float16 v16h;
typedef __attribute__((ext_vector_type(8)))  _Float16 v8h;
typedef __attribute__((ext_vector_type(8)))  float    v8f;

#define HIDDEN 128
#define KDIM   256
#define APAD   264                       // halves; row stride 528 B (16B aligned, bank-rotating)
#define SA_BYTES   (128 * APAD * 2)      // 67584 B activation stage
#define SPART_FLOATS (128 * 16)          // [edge][wave*2 + subgroup] partials
#define SMEM_BYTES (SA_BYTES + SPART_FLOATS * 4)

// Add the quad-mate's value selected by DPP quad_perm (pure VALU, no LDS traffic).
// 0xB1 = {1,0,3,2} (lane ^ 1), 0x4E = {2,3,0,1} (lane ^ 2)
template <int CTRL>
__device__ __forceinline__ float quad_xor_add(float x) {
    int y = __builtin_amdgcn_update_dpp(0, __float_as_int(x), CTRL, 0xf, 0xf, true);
    return x + __int_as_float(y);
}

__global__ __launch_bounds__(256) void edge_decoder_wmma(
    const float* __restrict__ z_src, const float* __restrict__ z_dst,
    const int* __restrict__ pos_src, const int* __restrict__ pos_dst,
    const int* __restrict__ neg_src, const int* __restrict__ neg_dst,
    const float* __restrict__ W1, const float* __restrict__ b1,
    const float* __restrict__ W2, const float* __restrict__ b2,
    float* __restrict__ out, int E, int ne, int nSuper)
{
    extern __shared__ char smem[];
    _Float16* sA    = (_Float16*)smem;             // [128 edges][APAD halves]
    float*    sPart = (float*)(smem + SA_BYTES);   // [128][16] per-edge partials

    const int tid  = threadIdx.x;
    const int lane = tid & 31;
    const int r    = lane & 15;    // row/col within 16
    const int hi   = lane >> 4;    // half-wave -> K phase
    const int hoff = hi * 8;       // K offset (halves)
    const int nt   = tid >> 5;     // wave id == owned N-tile (0..7)
    const int sub  = r >> 3;       // 8-lane subgroup id within half-wave

    // ---- Resident B fragments: wave nt owns hidden dims n = nt*16 + r, all 8 K-steps.
    const int ncol = nt * 16 + r;
    v16h bfrag[8];
#pragma unroll
    for (int kb = 0; kb < 8; ++kb) {
        const int kbase = kb * 32 + hoff;
        v16h bv;
#pragma unroll
        for (int q = 0; q < 8; ++q) {
            bv[q]     = (_Float16)W1[(kbase + q)      * HIDDEN + ncol];
            bv[8 + q] = (_Float16)W1[(kbase + 16 + q) * HIDDEN + ncol];
        }
        bfrag[kb] = bv;
    }

    const float b1r   = b1[ncol];
    const float w2r   = W2[ncol];
    const float bias2 = b2[0];

    // ---- Staging role: 2 threads per edge; half=0 -> z_src row, half=1 -> z_dst row
    const int eloc = tid >> 1;
    const int half = tid & 1;

    for (int st = blockIdx.x; st < nSuper; st += gridDim.x) {
        // ---- Gather + convert 128 rows (f32 -> f16) into LDS stage
        {
            const int e  = st * 128 + eloc;
            const int ec = (e < ne) ? e : 0;
            int idxv;
            if (ec < E) idxv = half ? pos_dst[ec]     : pos_src[ec];
            else        idxv = half ? neg_dst[ec - E] : neg_src[ec - E];
            const float* row = (half ? z_dst : z_src) + (long)idxv * HIDDEN;
            _Float16* dp = sA + eloc * APAD + half * 128;
#pragma unroll
            for (int j = 0; j < 16; ++j) {
                const float4 f0 = *(const float4*)(row + j * 8 + 0);
                const float4 f1 = *(const float4*)(row + j * 8 + 4);
                v8h hv;
                hv[0]=(_Float16)f0.x; hv[1]=(_Float16)f0.y; hv[2]=(_Float16)f0.z; hv[3]=(_Float16)f0.w;
                hv[4]=(_Float16)f1.x; hv[5]=(_Float16)f1.y; hv[6]=(_Float16)f1.z; hv[7]=(_Float16)f1.w;
                *(v8h*)(dp + j * 8) = hv;
            }
        }
        __syncthreads();

        // ---- 8 M-tiles: 16 batched ds_load_b128, sched barrier, 8 back-to-back wmma
#pragma unroll
        for (int mt = 0; mt < 8; ++mt) {
            const _Float16* ap = sA + (mt * 16 + r) * APAD + hoff;
            v8h alo[8], ahi[8];
#pragma unroll
            for (int kb = 0; kb < 8; ++kb) {          // all A loads issued up front
                alo[kb] = *(const v8h*)(ap + kb * 32);
                ahi[kb] = *(const v8h*)(ap + kb * 32 + 16);
            }
            __builtin_amdgcn_sched_barrier(0);        // keep loads ahead of the wmma chain

            v8f c;                                     // accumulator pre-seeded with b1 (N-const)
#pragma unroll
            for (int i = 0; i < 8; ++i) c[i] = b1r;
#pragma unroll
            for (int kb = 0; kb < 8; ++kb) {
                v16h af;
#pragma unroll
                for (int q = 0; q < 8; ++q) { af[q] = alo[kb][q]; af[8 + q] = ahi[kb][q]; }
                c = __builtin_amdgcn_wmma_f32_16x16x32_f16(
                        false, af, false, bfrag[kb], (short)0, c, false, false);
            }

            // ReLU + scale by W2
            float p[8];
#pragma unroll
            for (int i = 0; i < 8; ++i) {
                const float h = c[i] > 0.0f ? c[i] : 0.0f;
                p[i] = h * w2r;
            }
            // xor1 + xor2 via DPP quad_perm (VALU), xor4 via one bpermute stage
#pragma unroll
            for (int i = 0; i < 8; ++i) p[i] = quad_xor_add<0xB1>(p[i]);
#pragma unroll
            for (int i = 0; i < 8; ++i) p[i] = quad_xor_add<0x4E>(p[i]);
            {
                float t[8];
#pragma unroll
                for (int i = 0; i < 8; ++i) t[i] = __shfl_xor(p[i], 4, 32);
#pragma unroll
                for (int i = 0; i < 8; ++i) p[i] += t[i];
            }
            // lanes r=0 and r=8 hold 8-lane subgroup sums: plain stores, no atomics
            if ((r & 7) == 0) {
#pragma unroll
                for (int i = 0; i < 8; ++i)
                    sPart[(mt * 16 + hi * 8 + i) * 16 + nt * 2 + sub] = p[i];
            }
        }
        __syncthreads();

        // ---- Final: per edge, sum 16 contiguous partials (4x ds_load_b128)
        if (tid < 128) {
            const float4* q = (const float4*)&sPart[tid * 16];
            const float4 q0 = q[0], q1 = q[1], q2 = q[2], q3 = q[3];
            float s = ((q0.x + q0.y) + (q0.z + q0.w))
                    + ((q1.x + q1.y) + (q1.z + q1.w))
                    + ((q2.x + q2.y) + (q2.z + q2.w))
                    + ((q3.x + q3.y) + (q3.z + q3.w));
            const int eo = st * 128 + tid;
            if (eo < ne) out[eo] = s + bias2;
        }
        __syncthreads();   // protect sA / sPart reuse next iteration
    }
}

extern "C" void kernel_launch(void* const* d_in, const int* in_sizes, int n_in,
                              void* d_out, int out_size, void* d_ws, size_t ws_size,
                              hipStream_t stream) {
    const float* z_src   = (const float*)d_in[0];
    const float* z_dst   = (const float*)d_in[1];
    const int*   pos_src = (const int*)d_in[2];
    const int*   pos_dst = (const int*)d_in[3];
    const int*   neg_src = (const int*)d_in[4];
    const int*   neg_dst = (const int*)d_in[5];
    const float* W1 = (const float*)d_in[6];
    const float* b1 = (const float*)d_in[7];
    const float* W2 = (const float*)d_in[8];
    const float* b2 = (const float*)d_in[9];
    float* out = (float*)d_out;

    const int E  = in_sizes[2];
    const int ne = 2 * E;
    const int nSuper = (ne + 127) / 128;

    int blocks = nSuper;
    if (blocks > 1024) blocks = 1024;
    if (blocks < 1) blocks = 1;

    edge_decoder_wmma<<<blocks, 256, SMEM_BYTES, stream>>>(
        z_src, z_dst, pos_src, pos_dst, neg_src, neg_dst,
        W1, b1, W2, b2, out, E, ne, nSuper);
}
